// EGNN_vel_Aether_7215545057984
// MI455X (gfx1250) — matrix operand order repaired and checked by reading, not compiled, not run on hardware
//
#include <hip/hip_runtime.h>
#include <stdint.h>

#define NN   10000
#define EE   320000
#define HH   64
#define LL   4

// fragment-ordered bf16 weight blob: per layer 36 fragments (W1:20, W2:8, cW1:8),
// each fragment = 32 lanes x 16 u16 = 512 u16.
#define FRAGS_PER_LAYER 36
#define U16_PER_LAYER   (FRAGS_PER_LAYER * 512)   // 18432 u16 = 36864 B = 9216 dwords

typedef __attribute__((ext_vector_type(16))) __bf16 v16bf;
typedef __attribute__((ext_vector_type(8)))  float  v8f;
typedef __attribute__((ext_vector_type(4)))  unsigned int u32x4;
typedef __attribute__((ext_vector_type(8)))  int    i32x8;
typedef __attribute__((ext_vector_type(4)))  int    i32x4;

union AFrag { v16bf bf; __bf16 h[16]; unsigned short u[16]; uint4 q[2]; };
union CAcc  { v8f v; float f[8]; };

__device__ __forceinline__ float siluf(float x) { return x / (1.0f + __expf(-x)); }

// ---------------------------------------------------------------------------
// Weight prep: f32 -> bf16, fragment (B-matrix) order.
// B layout for v_wmma_f32_16x16x32_bf16: lane half h, element j -> K = h*8 + j (j < 8), 16 + h*8 + (j-8) (j >= 8),
// column N = lane & 15.
// ---------------------------------------------------------------------------
__global__ __launch_bounds__(256)
void prep_weights_kernel(const float* __restrict__ eW1,   // [L,131,64]
                         const float* __restrict__ eW2,   // [L,64,64]
                         const float* __restrict__ cW1,   // [L,64,64]
                         __bf16* __restrict__ eblob)
{
    int idx = blockIdx.x * 256 + threadIdx.x;
    if (idx >= LL * U16_PER_LAYER) return;
    int layer = idx / U16_PER_LAYER;
    int r     = idx % U16_PER_LAYER;
    int frag  = r / 512;
    int w     = r % 512;
    int lane  = w / 16;
    int j     = w % 16;
    int half  = lane >> 4;
    int n16   = lane & 15;
    int localK = (j < 8) ? (half * 8 + j) : (16 + half * 8 + (j - 8));   // B-matrix K mapping
    float val = 0.0f;
    if (frag < 20) {                             // W1: 5 k-tiles x 4 n-tiles
        int kt = frag / 4, nt = frag % 4;
        int K = kt * 32 + localK;
        if (K < 131) val = eW1[((size_t)layer * 131 + K) * 64 + nt * 16 + n16];
    } else if (frag < 28) {                      // W2: 2 k-tiles x 4 n-tiles
        int f = frag - 20; int kt = f / 4, nt = f % 4;
        int K = kt * 32 + localK;
        val = eW2[((size_t)layer * 64 + K) * 64 + nt * 16 + n16];
    } else {                                     // cW1: 2 k-tiles x 4 n-tiles
        int f = frag - 28; int kt = f / 4, nt = f % 4;
        int K = kt * 32 + localK;
        val = cW1[((size_t)layer * 64 + K) * 64 + nt * 16 + n16];
    }
    eblob[idx] = (__bf16)val;
}

// ---------------------------------------------------------------------------
// Node embedding + FieldNetwork + coord init
// ---------------------------------------------------------------------------
__global__ __launch_bounds__(256)
void embed_field_kernel(const float* __restrict__ h, const float* __restrict__ x,
                        const float* __restrict__ vel, const int* __restrict__ charges,
                        const float* __restrict__ embW, const float* __restrict__ embB,
                        const float* __restrict__ clsEmb,
                        const float* __restrict__ fW1, const float* __restrict__ fb1,
                        const float* __restrict__ fW2, const float* __restrict__ fb2,
                        const float* __restrict__ fW3, const float* __restrict__ fb3,
                        float* __restrict__ hh0, float* __restrict__ field,
                        float* __restrict__ coord)
{
    int n = blockIdx.x * blockDim.x + threadIdx.x;
    if (n >= NN) return;
    float hv = h[n];
    #pragma unroll
    for (int j = 0; j < 64; j++) hh0[(size_t)n * 64 + j] = hv * embW[j] + embB[j];

    float finp[22];
    #pragma unroll
    for (int d = 0; d < 3; d++) {
        float xv = x[n * 3 + d];
        finp[d]     = xv;
        finp[3 + d] = vel[n * 3 + d];
        coord[n * 3 + d] = xv;
    }
    int cls = charges[n];
    #pragma unroll
    for (int k = 0; k < 16; k++) finp[6 + k] = clsEmb[cls * 16 + k];

    float fh1[32];
    #pragma unroll
    for (int j = 0; j < 32; j++) {
        float a = fb1[j];
        #pragma unroll
        for (int k = 0; k < 22; k++) a += finp[k] * fW1[k * 32 + j];
        fh1[j] = siluf(a);
    }
    float fh2[32];
    #pragma unroll
    for (int j = 0; j < 32; j++) {
        float a = fb2[j];
        #pragma unroll
        for (int k = 0; k < 32; k++) a += fh1[k] * fW2[k * 32 + j];
        fh2[j] = siluf(a);
    }
    #pragma unroll
    for (int d = 0; d < 3; d++) {
        float a = fb3[d];
        #pragma unroll
        for (int k = 0; k < 32; k++) a += fh2[k] * fW3[k * 3 + d];
        field[n * 3 + d] = a;
    }
}

__global__ __launch_bounds__(256)
void zero_kernel(float* __restrict__ p, int count)
{
    int i = blockIdx.x * 256 + threadIdx.x;
    if (i < count) p[i] = 0.0f;
}

// ---------------------------------------------------------------------------
// Edge kernel: per wave one 16-edge tile; per block the layer's weight blob is
// staged into LDS (TDM tensor_load_to_lds when available, else cooperative copy).
//   GEMM1: e_in[16x160] x W1[160x64]   (bf16 WMMA, f32 acc) -> silu -> LDS
//   GEMM2: t1[16x64]    x W2[64x64]    -> silu = m -> magg scatter + LDS
//   GEMM3: m[16x64]     x cW1[64x64]   -> silu . cW2 -> cw -> csum/cnt scatter
// ---------------------------------------------------------------------------
__device__ __forceinline__ AFrag load_bfrag(const __bf16* base, int fragIdx, int lane)
{
    AFrag b;
    const uint4* p = (const uint4*)(base + fragIdx * 512 + lane * 16);
    b.q[0] = p[0];
    b.q[1] = p[1];
    return b;
}

#if __has_builtin(__builtin_amdgcn_tensor_load_to_lds) && __has_builtin(__builtin_amdgcn_s_wait_tensorcnt)
#define USE_TDM 1
#else
#define USE_TDM 0
#endif

__global__ __launch_bounds__(256)
void edge_kernel(int layer,
                 const int*   __restrict__ edges,      // [2,E]
                 const float* __restrict__ hh,         // [N,64]
                 const float* __restrict__ coord,      // [N,3]
                 const float* __restrict__ edge_attr,  // [E,2]
                 const __bf16* __restrict__ eblob,
                 const float* __restrict__ e_b1,       // [L,64]
                 const float* __restrict__ e_b2,       // [L,64]
                 const float* __restrict__ c_b1,       // [L,64]
                 const float* __restrict__ c_W2,       // [L,64]
                 float* __restrict__ csum,             // [N,3]
                 float* __restrict__ cnt,              // [N]
                 float* __restrict__ magg)             // [N,64]
{
    __shared__ __attribute__((aligned(16))) __bf16 sW[U16_PER_LAYER];   // 36 KB weight blob
    __shared__ __attribute__((aligned(16))) __bf16 sT[8][16][64];       // per-wave act tile
    __shared__ float sCw[8][16];

    const int lane = threadIdx.x & 31;
    const int wave = threadIdx.x >> 5;
    const int half = lane >> 4;
    const int l16  = lane & 15;
    const int e0   = (blockIdx.x * 8 + wave) * 16;
    const int my_e = e0 + l16;

    const __bf16* lb = eblob + (size_t)layer * U16_PER_LAYER;

    // ---- stage this layer's weight fragments into LDS -------------------
#if USE_TDM
    if (threadIdx.x < 32) {                       // one wave issues the DMA
        unsigned lds_off = (unsigned)(uintptr_t)(&sW[0]);
        unsigned long long ga = (unsigned long long)(uintptr_t)lb;
        const unsigned ndw = U16_PER_LAYER / 2;   // 9216 dwords
        u32x4 g0;
        g0[0] = 1u;                               // count=1 (valid descriptor)
        g0[1] = lds_off;                          // lds_addr
        g0[2] = (unsigned)(ga & 0xffffffffu);     // global_addr[31:0]
        g0[3] = (unsigned)((ga >> 32) & 0x1ffffffu) | (2u << 30); // addr[56:32], type=2
        i32x8 g1;
        g1[0] = (int)(2u << 16);                  // data_size = 4B, no multicast/flags
        g1[1] = (int)((ndw & 0xffffu) << 16);     // tensor_dim0[15:0] at bits[63:48]
        g1[2] = (int)((ndw >> 16) | (1u << 16));  // tensor_dim0[31:16], tensor_dim1=1
        g1[3] = (int)(ndw << 16);                 // tensor_dim1 hi=0, tile_dim0 = 9216
        g1[4] = 0;                                // tile_dim1/tile_dim2 unused (1-D tile)
        g1[5] = (int)ndw;                         // tensor_dim0_stride low
        g1[6] = 0;
        g1[7] = 0;
        i32x4 z4 = {0, 0, 0, 0};                  // groups 2/3 unused (<=2-D tensor)
        i32x8 z8 = {0, 0, 0, 0, 0, 0, 0, 0};
        __builtin_amdgcn_tensor_load_to_lds(g0, g1, z4, z4, z8, 0);
        __builtin_amdgcn_s_wait_tensorcnt(0);
    }
#else
    {
        const uint32_t* gsrc = (const uint32_t*)lb;
        uint32_t* ldst = (uint32_t*)&sW[0];
        #pragma unroll
        for (int t = 0; t < U16_PER_LAYER / 2 / 256; t++)
            ldst[threadIdx.x + t * 256] = gsrc[threadIdx.x + t * 256];
    }
#endif
    __syncthreads();

    const int* edges0 = edges;
    const int* edges1 = edges + EE;
    const int row = edges0[my_e];
    const int col = edges1[my_e];

    float dx = coord[row * 3 + 0] - coord[col * 3 + 0];
    float dy = coord[row * 3 + 1] - coord[col * 3 + 1];
    float dz = coord[row * 3 + 2] - coord[col * 3 + 2];
    float radial = dx * dx + dy * dy + dz * dz;
    float ea0 = edge_attr[(size_t)my_e * 2 + 0];
    float ea1 = edge_attr[(size_t)my_e * 2 + 1];

    // ---- A fragments for GEMM1 (5 k-tiles of 32 features). A layout:
    //   lane half h: elems 0..7 -> K = kt*32 + h*8 + j ; elems 8..15 -> K = kt*32 + 16 + h*8 + (j-8)
    AFrag a1[5];
    #pragma unroll
    for (int kt = 0; kt < 4; kt++) {
        const float* src = (kt < 2) ? (hh + (size_t)row * 64 + kt * 32)
                                    : (hh + (size_t)col * 64 + (kt - 2) * 32);
        const float4* p0 = (const float4*)(src + half * 8);
        const float4* p1 = (const float4*)(src + 16 + half * 8);
        float4 f0 = p0[0], f1 = p0[1], f2 = p1[0], f3 = p1[1];
        a1[kt].h[0]  = (__bf16)f0.x; a1[kt].h[1]  = (__bf16)f0.y;
        a1[kt].h[2]  = (__bf16)f0.z; a1[kt].h[3]  = (__bf16)f0.w;
        a1[kt].h[4]  = (__bf16)f1.x; a1[kt].h[5]  = (__bf16)f1.y;
        a1[kt].h[6]  = (__bf16)f1.z; a1[kt].h[7]  = (__bf16)f1.w;
        a1[kt].h[8]  = (__bf16)f2.x; a1[kt].h[9]  = (__bf16)f2.y;
        a1[kt].h[10] = (__bf16)f2.z; a1[kt].h[11] = (__bf16)f2.w;
        a1[kt].h[12] = (__bf16)f3.x; a1[kt].h[13] = (__bf16)f3.y;
        a1[kt].h[14] = (__bf16)f3.z; a1[kt].h[15] = (__bf16)f3.w;
    }
    #pragma unroll
    for (int j = 0; j < 16; j++) a1[4].u[j] = 0;
    if (half == 0) {   // K=128 radial, 129/130 edge_attr, rest zero pad
        a1[4].h[0] = (__bf16)radial;
        a1[4].h[1] = (__bf16)ea0;
        a1[4].h[2] = (__bf16)ea1;
    }

    // ---- GEMM1 -> silu -> sT (bf16)
    #pragma unroll
    for (int nt = 0; nt < 4; nt++) {
        CAcc c;
        #pragma unroll
        for (int r = 0; r < 8; r++) c.f[r] = 0.0f;
        #pragma unroll
        for (int kt = 0; kt < 5; kt++) {
            AFrag b = load_bfrag(sW, kt * 4 + nt, lane);
            c.v = __builtin_amdgcn_wmma_f32_16x16x32_bf16(false, a1[kt].bf, false, b.bf,
                                                          (short)0, c.v, false, false);
        }
        const float b1 = e_b1[layer * 64 + nt * 16 + l16];
        #pragma unroll
        for (int r = 0; r < 8; r++) {
            float v = siluf(c.f[r] + b1);
            sT[wave][r + half * 8][nt * 16 + l16] = (__bf16)v;   // (M, N)
        }
    }
    __syncthreads();

    // row index per output-row M (for scatters from C layout)
    int rowM[8];
    #pragma unroll
    for (int r = 0; r < 8; r++) rowM[r] = edges0[e0 + half * 8 + r];

    // ---- GEMM2 (t1 x W2) -> silu = m ; scatter magg ; keep m in regs
    const uint4* tp = (const uint4*)(&sT[wave][l16][0]);       // row M=l16, 8 uint4 per row
    AFrag a2[2];
    #pragma unroll
    for (int kt = 0; kt < 2; kt++) {
        a2[kt].q[0] = tp[(kt * 32 + half * 8) >> 3];
        a2[kt].q[1] = tp[(kt * 32 + 16 + half * 8) >> 3];
    }
    float mreg[4][8];
    #pragma unroll
    for (int nt = 0; nt < 4; nt++) {
        CAcc c;
        #pragma unroll
        for (int r = 0; r < 8; r++) c.f[r] = 0.0f;
        #pragma unroll
        for (int kt = 0; kt < 2; kt++) {
            AFrag b = load_bfrag(sW, 20 + kt * 4 + nt, lane);
            c.v = __builtin_amdgcn_wmma_f32_16x16x32_bf16(false, a2[kt].bf, false, b.bf,
                                                          (short)0, c.v, false, false);
        }
        const float b2 = e_b2[layer * 64 + nt * 16 + l16];
        #pragma unroll
        for (int r = 0; r < 8; r++) {
            float v = siluf(c.f[r] + b2);
            mreg[nt][r] = v;
            atomicAdd(&magg[(size_t)rowM[r] * 64 + nt * 16 + l16], v);
        }
    }
    __syncthreads();

    // overwrite sT with m (bf16) for GEMM3's A
    #pragma unroll
    for (int nt = 0; nt < 4; nt++)
        #pragma unroll
        for (int r = 0; r < 8; r++)
            sT[wave][r + half * 8][nt * 16 + l16] = (__bf16)mreg[nt][r];
    __syncthreads();

    AFrag a3[2];
    #pragma unroll
    for (int kt = 0; kt < 2; kt++) {
        a3[kt].q[0] = tp[(kt * 32 + half * 8) >> 3];
        a3[kt].q[1] = tp[(kt * 32 + 16 + half * 8) >> 3];
    }

    // ---- GEMM3 (m x cW1) -> silu . cW2 -> per-edge coord weight
    float p[8];
    #pragma unroll
    for (int r = 0; r < 8; r++) p[r] = 0.0f;
    #pragma unroll
    for (int nt = 0; nt < 4; nt++) {
        CAcc c;
        #pragma unroll
        for (int r = 0; r < 8; r++) c.f[r] = 0.0f;
        #pragma unroll
        for (int kt = 0; kt < 2; kt++) {
            AFrag b = load_bfrag(sW, 28 + kt * 4 + nt, lane);
            c.v = __builtin_amdgcn_wmma_f32_16x16x32_bf16(false, a3[kt].bf, false, b.bf,
                                                          (short)0, c.v, false, false);
        }
        const float bc = c_b1[layer * 64 + nt * 16 + l16];
        const float w2 = c_W2[layer * 64 + nt * 16 + l16];
        #pragma unroll
        for (int r = 0; r < 8; r++) p[r] += siluf(c.f[r] + bc) * w2;
    }
    // reduce over the 16 lanes of each half (masks <16 keep halves separate)
    #pragma unroll
    for (int r = 0; r < 8; r++) {
        p[r] += __shfl_xor(p[r], 1);
        p[r] += __shfl_xor(p[r], 2);
        p[r] += __shfl_xor(p[r], 4);
        p[r] += __shfl_xor(p[r], 8);
    }
    if (l16 == 0) {
        #pragma unroll
        for (int r = 0; r < 8; r++) sCw[wave][half * 8 + r] = p[r];
    }
    __syncthreads();
    if (half == 0) {
        float cw = sCw[wave][l16];
        atomicAdd(&csum[row * 3 + 0], dx * cw);
        atomicAdd(&csum[row * 3 + 1], dy * cw);
        atomicAdd(&csum[row * 3 + 2], dz * cw);
        atomicAdd(&cnt[row], 1.0f);
    }
}

// ---------------------------------------------------------------------------
// Node-side kernels (VALU; negligible FLOPs)
// ---------------------------------------------------------------------------
__global__ __launch_bounds__(256)
void node_hidden_kernel(const float* __restrict__ hh, const float* __restrict__ magg,
                        const float* __restrict__ nW1, const float* __restrict__ nb1,
                        const float* __restrict__ vW1, const float* __restrict__ vb1,
                        const float* __restrict__ gW1, const float* __restrict__ gb1,
                        float* __restrict__ tN, float* __restrict__ tV, float* __restrict__ tG)
{
    int idx = blockIdx.x * blockDim.x + threadIdx.x;
    if (idx >= NN * 64) return;
    int n = idx >> 6, j = idx & 63;
    float aN = nb1[j], aV = vb1[j], aG = gb1[j];
    const float* hr = hh + (size_t)n * 64;
    #pragma unroll 4
    for (int k = 0; k < 64; k++) {
        float hv = hr[k];
        aN += hv * nW1[k * 64 + j];
        aV += hv * vW1[k * 64 + j];
        aG += hv * gW1[k * 64 + j];
    }
    const float* mr = magg + (size_t)n * 64;
    #pragma unroll 4
    for (int k = 0; k < 64; k++) aN += mr[k] * nW1[(64 + k) * 64 + j];
    tN[idx] = siluf(aN);
    tV[idx] = siluf(aV);
    tG[idx] = siluf(aG);
}

__global__ __launch_bounds__(256)
void node_coord_kernel(const float* __restrict__ tV, const float* __restrict__ tG,
                       const float* __restrict__ vW2, const float* __restrict__ vb2,
                       const float* __restrict__ gW2, const float* __restrict__ gb2,
                       const float* __restrict__ csum, const float* __restrict__ cnt,
                       const float* __restrict__ vel, const float* __restrict__ field,
                       float* __restrict__ coord, float* __restrict__ outc)
{
    int n = blockIdx.x * blockDim.x + threadIdx.x;
    if (n >= NN) return;
    float vw = vb2[0], gw = gb2[0];
    const float* tv = tV + (size_t)n * 64;
    const float* tg = tG + (size_t)n * 64;
    #pragma unroll 8
    for (int j = 0; j < 64; j++) {
        vw += tv[j] * vW2[j];
        gw += tg[j] * gW2[j];
    }
    float cd = fmaxf(cnt[n], 1.0f);
    #pragma unroll
    for (int d = 0; d < 3; d++) {
        float c = coord[n * 3 + d] + csum[n * 3 + d] / cd
                + vw * vel[n * 3 + d] + gw * field[n * 3 + d];
        coord[n * 3 + d] = c;
        outc[n * 3 + d]  = c;
    }
}

__global__ __launch_bounds__(256)
void node_out_kernel(const float* __restrict__ tN, const float* __restrict__ nW2,
                     const float* __restrict__ nb2, float* __restrict__ hh_out)
{
    int idx = blockIdx.x * blockDim.x + threadIdx.x;
    if (idx >= NN * 64) return;
    int n = idx >> 6, j = idx & 63;
    float a = nb2[j];
    const float* tr = tN + (size_t)n * 64;
    #pragma unroll 4
    for (int k = 0; k < 64; k++) a += tr[k] * nW2[k * 64 + j];
    hh_out[idx] = a;
}

// ---------------------------------------------------------------------------
extern "C" void kernel_launch(void* const* d_in, const int* in_sizes, int n_in,
                              void* d_out, int out_size, void* d_ws, size_t ws_size,
                              hipStream_t stream)
{
    const float* h         = (const float*)d_in[0];
    const float* x         = (const float*)d_in[1];
    const float* vel       = (const float*)d_in[2];
    const float* edge_attr = (const float*)d_in[3];
    const int*   edges     = (const int*)  d_in[4];
    const int*   charges   = (const int*)  d_in[5];
    const float* emb_W     = (const float*)d_in[6];
    const float* emb_b     = (const float*)d_in[7];
    const float* cls_emb   = (const float*)d_in[8];
    const float* f_W1 = (const float*)d_in[9];
    const float* f_b1 = (const float*)d_in[10];
    const float* f_W2 = (const float*)d_in[11];
    const float* f_b2 = (const float*)d_in[12];
    const float* f_W3 = (const float*)d_in[13];
    const float* f_b3 = (const float*)d_in[14];
    const float* e_W1 = (const float*)d_in[15];
    const float* e_b1 = (const float*)d_in[16];
    const float* e_W2 = (const float*)d_in[17];
    const float* e_b2 = (const float*)d_in[18];
    const float* n_W1 = (const float*)d_in[19];
    const float* n_b1 = (const float*)d_in[20];
    const float* n_W2 = (const float*)d_in[21];
    const float* n_b2 = (const float*)d_in[22];
    const float* c_W1 = (const float*)d_in[23];
    const float* c_b1 = (const float*)d_in[24];
    const float* c_W2 = (const float*)d_in[25];
    const float* v_W1 = (const float*)d_in[26];
    const float* v_b1 = (const float*)d_in[27];
    const float* v_W2 = (const float*)d_in[28];
    const float* v_b2 = (const float*)d_in[29];
    const float* g_W1 = (const float*)d_in[30];
    const float* g_b1 = (const float*)d_in[31];
    const float* g_W2 = (const float*)d_in[32];
    const float* g_b2 = (const float*)d_in[33];

    char* ws = (char*)d_ws;
    size_t off = 0;
    auto balloc = [&](size_t bytes) -> char* {
        char* p = ws + off;
        off = (off + bytes + 255) & ~(size_t)255;
        return p;
    };
    __bf16* eblob = (__bf16*)balloc((size_t)LL * U16_PER_LAYER * 2);
    float* hhA   = (float*)balloc((size_t)NN * 64 * 4);
    float* hhB   = (float*)balloc((size_t)NN * 64 * 4);
    float* field = (float*)balloc((size_t)NN * 3 * 4);
    float* coord = (float*)balloc((size_t)NN * 3 * 4);
    float* csum  = (float*)balloc((size_t)NN * 68 * 4);   // csum|cnt|magg packed
    float* cnt   = csum + (size_t)NN * 3;
    float* magg  = cnt  + (size_t)NN;
    float* tN    = (float*)balloc((size_t)NN * 64 * 4);
    float* tV    = (float*)balloc((size_t)NN * 64 * 4);
    float* tG    = (float*)balloc((size_t)NN * 64 * 4);

    prep_weights_kernel<<<(LL * U16_PER_LAYER + 255) / 256, 256, 0, stream>>>(e_W1, e_W2, c_W1, eblob);
    embed_field_kernel<<<(NN + 255) / 256, 256, 0, stream>>>(
        h, x, vel, charges, emb_W, emb_b, cls_emb,
        f_W1, f_b1, f_W2, f_b2, f_W3, f_b3, hhA, field, coord);

    for (int i = 0; i < LL; i++) {
        const float* hin  = (i & 1) ? hhB : hhA;
        float*       hout = (i & 1) ? hhA : hhB;
        zero_kernel<<<(NN * 68 + 255) / 256, 256, 0, stream>>>(csum, NN * 68);
        edge_kernel<<<EE / 16 / 8, 256, 0, stream>>>(
            i, edges, hin, coord, edge_attr, eblob,
            e_b1, e_b2, c_b1, c_W2, csum, cnt, magg);
        node_hidden_kernel<<<(NN * 64) / 256, 256, 0, stream>>>(
            hin, magg,
            n_W1 + (size_t)i * 128 * 64, n_b1 + i * 64,
            v_W1 + (size_t)i * 64 * 64,  v_b1 + i * 64,
            g_W1 + (size_t)i * 64 * 64,  g_b1 + i * 64,
            tN, tV, tG);
        node_coord_kernel<<<(NN + 255) / 256, 256, 0, stream>>>(
            tV, tG, v_W2 + i * 64, v_b2 + i, g_W2 + i * 64, g_b2 + i,
            csum, cnt, vel, field, coord, (float*)d_out);
        node_out_kernel<<<(NN * 64) / 256, 256, 0, stream>>>(
            tN, n_W2 + (size_t)i * 64 * 64, n_b2 + i * 64, hout);
    }
}
